// MultiThreadMemory_8194797601292
// MI455X (gfx1250) — matrix operand order repaired and checked by reading, not compiled, not run on hardware
//
#include <hip/hip_runtime.h>
#include <math.h>

typedef _Float16 f16;
typedef __attribute__((ext_vector_type(16))) _Float16 v16h;
typedef __attribute__((ext_vector_type(8)))  _Float16 v8h;
typedef __attribute__((ext_vector_type(8)))  float    v8f;
typedef __attribute__((ext_vector_type(4)))  float    v4f;

#define NB   4
#define NSQ  1024
#define NSKV 4096
#define DM   512
#define HD   8
#define DK   64
#define KSEL 32   // TOPK
#define CHUNK 512

// ---------------------------------------------------------------------------
// WMMA fragment helpers (contiguous k order chosen by this code, not the ISA table order; wave32)
// A (16x32 f16): lane L<16 -> M=L, halfs = K[0..15] contiguous;
//                lane L>=16 -> M=L-16, halfs = K[16..31] contiguous
// B (32x16 f16): lane L<16 -> N=L, halfs = K[0..15] contiguous;
//                lane L>=16 -> N=L-16, halfs = K[16..31] contiguous
// C/D (16x16 f32): lane L<16 -> N=L, reg r -> M=r; lane>=16 -> M=8+r
// ---------------------------------------------------------------------------
__device__ inline v16h load_a_frag(const f16* __restrict__ rowp, int hsel) {
  return *reinterpret_cast<const v16h*>(rowp + hsel * 16);
}

__device__ inline v16h load_b_frag(const f16* __restrict__ p) {
  return *reinterpret_cast<const v16h*>(p);  // 32B, 32B-aligned
}

__device__ inline v8f wmma_f16(v16h a, v16h b, v8f c) {
  return __builtin_amdgcn_wmma_f32_16x16x32_f16(
      /*neg_a=*/false, a, /*neg_b=*/false, b,
      /*c_mod=*/(short)0, c, /*reuse_a=*/false, /*reuse_b=*/false);
}

// ---------------------------------------------------------------------------
// Wt[n*512 + k] = (f16) W[k*512 + n]
// ---------------------------------------------------------------------------
__global__ void prep_weight(const float* __restrict__ W, f16* __restrict__ Wt) {
  int k = blockIdx.x * 16 + threadIdx.x;
  int n = blockIdx.y * 16 + threadIdx.y;
  Wt[(size_t)n * DM + k] = (f16)W[(size_t)k * DM + n];
}

__global__ void f32_to_f16(const float* __restrict__ in, f16* __restrict__ out, int n) {
  int i = blockIdx.x * blockDim.x + threadIdx.x;
  int stride = gridDim.x * blockDim.x;
  for (; i < n; i += stride) out[i] = (f16)in[i];
}

// ---------------------------------------------------------------------------
// out[M,512] = A_f16[M,512] @ Wt_f16^T + bias   (Wt stored [N][K])
// block = 256 threads = 8 waves; block tile 32(M) x 64(N); wave tile 16x16
// ---------------------------------------------------------------------------
__global__ __launch_bounds__(256) void gemm512_f16(
    const f16* __restrict__ A, const f16* __restrict__ Wt,
    const float* __restrict__ bias, float* __restrict__ out) {
  const int tid = threadIdx.x, w = tid >> 5, lane = tid & 31;
  const int mlane = lane & 15, hsel = lane >> 4;
  const int rowbase = blockIdx.y * 32 + (w & 1) * 16;
  const int colbase = blockIdx.x * 64 + (w >> 1) * 16;
  const f16* arow = A  + (size_t)(rowbase + mlane) * DM;
  const f16* brow = Wt + (size_t)(colbase + mlane) * DM;
  v8f c = {};
#pragma unroll
  for (int kk = 0; kk < DM; kk += 32) {
    v16h a = load_a_frag(arow + kk, hsel);
    v16h b = load_b_frag(brow + kk + hsel * 16);
    c = wmma_f16(a, b, c);
  }
  const int n = colbase + mlane;
  const float bv = bias[n];
  const int mbase = rowbase + hsel * 8;
#pragma unroll
  for (int r = 0; r < 8; ++r)
    out[(size_t)(mbase + r) * DM + n] = c[r] + bv;
}

// ---------------------------------------------------------------------------
// L2-normalize each 64-wide head chunk; emit head-major f16 [B,H,S,64]
// one wave32 per chunk
// ---------------------------------------------------------------------------
__global__ __launch_bounds__(256) void l2norm_split(
    const float* __restrict__ proj, f16* __restrict__ outh, int S) {
  const int widx = blockIdx.x * 8 + (threadIdx.x >> 5);
  const int lane = threadIdx.x & 31;
  const int h = widx % HD;
  const int rem = widx / HD;
  const int s = rem % S;
  const int b = rem / S;
  const float2 xv = *(const float2*)(proj + ((size_t)(b * S + s)) * DM + h * DK + lane * 2);
  float ss = xv.x * xv.x + xv.y * xv.y;
#pragma unroll
  for (int off = 16; off > 0; off >>= 1) ss += __shfl_xor(ss, off, 32);
  const float inv = 1.0f / fmaxf(sqrtf(ss), 1e-12f);
  f16* q = outh + (((size_t)(b * HD + h)) * S + s) * DK + lane * 2;
  q[0] = (f16)(xv.x * inv);
  q[1] = (f16)(xv.y * inv);
}

// ---------------------------------------------------------------------------
// Scores (WMMA) -> top-32 -> softmax -> gather V -> ctx, idx
// block = 128 threads (4 waves), one block per (b, h, 16-query tile)
// ---------------------------------------------------------------------------
__global__ __launch_bounds__(128) void attn_topk(
    const f16* __restrict__ qh, const f16* __restrict__ kh,
    const float* __restrict__ vproj, float* __restrict__ x,
    int* __restrict__ idx_out) {
  __shared__ float sc[16][CHUNK + 4];
  __shared__ float topv[16][KSEL];
  __shared__ int   topi[16][KSEL];
  __shared__ float pbuf[16][KSEL];
  __shared__ int   cnt[16];

  const int tid = threadIdx.x, w = tid >> 5, lane = tid & 31;
  const int mlane = lane & 15, hsel = lane >> 4;
  const int qtile = blockIdx.x, h = blockIdx.y, b = blockIdx.z;
  const size_t qrow0 = ((size_t)(b * HD + h)) * NSQ + qtile * 16;
  const size_t krow0 = ((size_t)(b * HD + h)) * NSKV;

  if (tid < 16) cnt[tid] = 0;

  // Q fragments: A operand covering K = 0..63 (two 16x32 fragments)
  const f16* qrow = qh + (qrow0 + mlane) * DK;
  const v16h a0 = load_a_frag(qrow, hsel);
  const v16h a1 = load_a_frag(qrow + 32, hsel);
  __syncthreads();

  for (int ch = 0; ch < NSKV; ch += CHUNK) {
    for (int t = w; t < CHUNK / 16; t += 4) {
      const f16* krow = kh + (krow0 + ch + t * 16 + mlane) * DK;
      v16h b0 = load_b_frag(krow + hsel * 16);
      v16h b1 = load_b_frag(krow + 32 + hsel * 16);
      v8f c = {};
      c = wmma_f16(a0, b0, c);
      c = wmma_f16(a1, b1, c);
      const int kl = t * 16 + mlane;
#pragma unroll
      for (int r = 0; r < 8; ++r)
        sc[hsel * 8 + r][kl] = c[r] * 0.125f;  // 1/sqrt(64)
    }
    __syncthreads();

    if (tid < 16) {  // maintain sorted (desc) top-32 per query row
      const int m = tid;
      int c0 = cnt[m];
      for (int k = 0; k < CHUNK; ++k) {
        float s = sc[m][k];
        if (c0 == KSEL && s <= topv[m][KSEL - 1]) continue;
        int top = (c0 < KSEL) ? c0 : KSEL - 1;
        int p2 = top;
        while (p2 > 0 && topv[m][p2 - 1] < s) --p2;  // equal keeps earlier index
        for (int j = top; j > p2; --j) {
          topv[m][j] = topv[m][j - 1];
          topi[m][j] = topi[m][j - 1];
        }
        topv[m][p2] = s;
        topi[m][p2] = ch + k;
        if (c0 < KSEL) ++c0;
      }
      cnt[m] = c0;
    }
    __syncthreads();
  }

  // softmax over sorted top-32 (T=1) + write indices
  if (tid < 16) {
    const int m = tid;
    const float mx = topv[m][0];
    float e[KSEL], sum = 0.f;
#pragma unroll
    for (int j = 0; j < KSEL; ++j) { e[j] = __expf(topv[m][j] - mx); sum += e[j]; }
    const float r = 1.0f / sum;
    int* op = idx_out + (qrow0 + m) * KSEL;
#pragma unroll
    for (int j = 0; j < KSEL; ++j) { pbuf[m][j] = e[j] * r; op[j] = topi[m][j]; }
  }
  __syncthreads();

  // ctx: 8 threads per query row, 8 d-values each; gather V at head offset
  const int m = tid >> 3, d0 = (tid & 7) * 8;
  float acc[8] = {};
  const float* vbase = vproj + (size_t)b * NSKV * DM + h * DK + d0;
  for (int j = 0; j < KSEL; ++j) {
    const float p = pbuf[m][j];
    const float* vp = vbase + (size_t)topi[m][j] * DM;
    v4f u0 = *(const v4f*)vp;
    v4f u1 = *(const v4f*)(vp + 4);
#pragma unroll
    for (int i = 0; i < 4; ++i) { acc[i] += p * u0[i]; acc[4 + i] += p * u1[i]; }
  }
  float* xp = x + ((size_t)b * NSQ + qtile * 16 + m) * DM + h * DK + d0;
#pragma unroll
  for (int i = 0; i < 8; ++i) xp[i] = acc[i];
}

// ---------------------------------------------------------------------------
extern "C" void kernel_launch(void* const* d_in, const int* in_sizes, int n_in,
                              void* d_out, int out_size, void* d_ws, size_t ws_size,
                              hipStream_t stream) {
  const float* query = (const float*)d_in[0];
  const float* key   = (const float*)d_in[1];
  const float* value = (const float*)d_in[2];
  const float* Wq = (const float*)d_in[3];  const float* bq = (const float*)d_in[4];
  const float* Wk = (const float*)d_in[5];  const float* bk = (const float*)d_in[6];
  const float* Wv = (const float*)d_in[7];  const float* bv = (const float*)d_in[8];
  const float* Wo = (const float*)d_in[9];  const float* bo = (const float*)d_in[10];

  float* out     = (float*)d_out;                       // [4,1024,512] f32
  int*   idx_out = (int*)d_out + NB * NSQ * DM;         // [4,8,1024,32] i32

  char* ws = (char*)d_ws;
  size_t off = 0;
  auto take = [&](size_t bytes) -> char* {
    char* p = ws + off;
    off += (bytes + 255) & ~(size_t)255;
    return p;
  };
  f16* Wt_q   = (f16*)take((size_t)DM * DM * sizeof(f16));
  f16* Wt_k   = (f16*)take((size_t)DM * DM * sizeof(f16));
  f16* Wt_v   = (f16*)take((size_t)DM * DM * sizeof(f16));
  f16* Wt_o   = (f16*)take((size_t)DM * DM * sizeof(f16));
  f16* act16  = (f16*)take((size_t)NB * NSKV * DM * sizeof(f16));   // reused staging
  float* kproj = (float*)take((size_t)NB * NSKV * DM * sizeof(float));
  float* vproj = (float*)take((size_t)NB * NSKV * DM * sizeof(float));
  float* qproj = (float*)take((size_t)NB * NSQ * DM * sizeof(float)); // reused as x
  f16* q_h    = (f16*)take((size_t)NB * HD * NSQ * DK * sizeof(f16));
  f16* k_h    = (f16*)take((size_t)NB * HD * NSKV * DK * sizeof(f16));
  (void)ws_size; (void)in_sizes; (void)n_in; (void)out_size;

  const dim3 wblk(16, 16), wgrd(DM / 16, DM / 16);
  prep_weight<<<wgrd, wblk, 0, stream>>>(Wq, Wt_q);
  prep_weight<<<wgrd, wblk, 0, stream>>>(Wk, Wt_k);
  prep_weight<<<wgrd, wblk, 0, stream>>>(Wv, Wt_v);
  prep_weight<<<wgrd, wblk, 0, stream>>>(Wo, Wt_o);

  const int Mq = NB * NSQ, Mkv = NB * NSKV;

  // K path
  f32_to_f16<<<1024, 256, 0, stream>>>(key, act16, Mkv * DM);
  gemm512_f16<<<dim3(DM / 64, Mkv / 32), 256, 0, stream>>>(act16, Wt_k, bk, kproj);
  l2norm_split<<<NB * NSKV, 256, 0, stream>>>(kproj, k_h, NSKV);

  // V path (stays f32, head offset used at gather time)
  f32_to_f16<<<1024, 256, 0, stream>>>(value, act16, Mkv * DM);
  gemm512_f16<<<dim3(DM / 64, Mkv / 32), 256, 0, stream>>>(act16, Wt_v, bv, vproj);

  // Q path
  f32_to_f16<<<1024, 256, 0, stream>>>(query, act16, Mq * DM);
  gemm512_f16<<<dim3(DM / 64, Mq / 32), 256, 0, stream>>>(act16, Wt_q, bq, qproj);
  l2norm_split<<<NB * NSQ, 256, 0, stream>>>(qproj, q_h, NSQ);

  // attention: scores (WMMA) + top-32 + softmax + gather; x reuses qproj buffer
  attn_topk<<<dim3(NSQ / 16, HD, NB), 128, 0, stream>>>(q_h, k_h, vproj, qproj, idx_out);

  // output projection
  f32_to_f16<<<1024, 256, 0, stream>>>(qproj, act16, Mq * DM);
  gemm512_f16<<<dim3(DM / 64, Mq / 32), 256, 0, stream>>>(act16, Wt_o, bo, out);
}
